// MultiHeadFlashSelfAttention_24635932410147
// MI455X (gfx1250) — compile-verified, matrix operand order checked by reading, not run on hardware
//
#include <hip/hip_runtime.h>
#include <hip/hip_bf16.h>

typedef __bf16 bf16_t;
typedef __attribute__((ext_vector_type(16))) __bf16 v16bf;
typedef __attribute__((ext_vector_type(8)))  float  v8f;
typedef __attribute__((ext_vector_type(4)))  int    v4i;

typedef __attribute__((address_space(1))) v4i* g_v4i_ptr;   // global int4*
typedef __attribute__((address_space(3))) v4i* l_v4i_ptr;   // LDS int4*

#if __has_builtin(__builtin_amdgcn_global_load_async_to_lds_b128) && \
    __has_builtin(__builtin_amdgcn_s_wait_asynccnt)
#define USE_ASYNC_LDS 1
#else
#define USE_ASYNC_LDS 0
#endif

static constexpr int Bc  = 4;
static constexpr int Sc  = 2048;
static constexpr int Dc  = 1024;
static constexpr int Hc  = 16;
static constexpr int DHc = 64;

// ---------------------------------------------------------------------------
// 16-lane-group max reduction via DPP butterflies (VALU pipe, no LDS traffic)
// ---------------------------------------------------------------------------
__device__ inline float dpp_max16(float x) {
    int t;
    t = __builtin_amdgcn_update_dpp(0, __builtin_bit_cast(int, x),
                                    0xB1 /*quad_perm(1,0,3,2)*/, 0xF, 0xF, true);
    x = fmaxf(x, __builtin_bit_cast(float, t));
    t = __builtin_amdgcn_update_dpp(0, __builtin_bit_cast(int, x),
                                    0x4E /*quad_perm(2,3,0,1)*/, 0xF, 0xF, true);
    x = fmaxf(x, __builtin_bit_cast(float, t));
    t = __builtin_amdgcn_update_dpp(0, __builtin_bit_cast(int, x),
                                    0x141 /*row_half_mirror*/, 0xF, 0xF, true);
    x = fmaxf(x, __builtin_bit_cast(float, t));
    t = __builtin_amdgcn_update_dpp(0, __builtin_bit_cast(int, x),
                                    0x140 /*row_mirror*/, 0xF, 0xF, true);
    x = fmaxf(x, __builtin_bit_cast(float, t));
    return x;
}

// ---------------------------------------------------------------------------
// fp32 -> bf16 cast (4 elements / thread)
// ---------------------------------------------------------------------------
__global__ void cast_f32_bf16(const float* __restrict__ src,
                              bf16_t* __restrict__ dst, int n) {
    int i = (blockIdx.x * blockDim.x + threadIdx.x) * 4;
    if (i + 3 < n) {
        float4 v = *(const float4*)(src + i);
        dst[i + 0] = (bf16_t)v.x;
        dst[i + 1] = (bf16_t)v.y;
        dst[i + 2] = (bf16_t)v.z;
        dst[i + 3] = (bf16_t)v.w;
    }
}

// ---------------------------------------------------------------------------
// RoPE (in place, bf16 head-major tensor [b*H+h][s][dh])
// ---------------------------------------------------------------------------
__global__ void rope_kernel(bf16_t* __restrict__ T,
                            const int* __restrict__ pos) {
    int pi  = blockIdx.x * blockDim.x + threadIdx.x;   // B*H*S*32 pairs
    int d2  = pi & 31;
    int row = pi >> 5;                                 // (b*H+h)*S + s
    int s   = row & (Sc - 1);
    float p   = (float)pos[s];
    float inv = __powf(10000.0f, -(float)(2 * d2) * (1.0f / 64.0f));
    float ang = p * inv;
    float cs, sn;
    __sincosf(ang, &sn, &cs);
    size_t base = (size_t)row * DHc + 2 * d2;
    float t1 = (float)T[base];
    float t2 = (float)T[base + 1];
    T[base]     = (bf16_t)(t1 * cs - t2 * sn);
    T[base + 1] = (bf16_t)(t1 * sn + t2 * cs);
}

// ---------------------------------------------------------------------------
// WMMA GEMM: C[8192][1024] = A[8192][1024] x W[1024][1024]^T (bf16, f32 acc)
// MODE 0: store bf16 head-major [b][h][s][dh];  MODE 1: store f32 [m][n]
// ---------------------------------------------------------------------------
template <int MODE>
__global__ __launch_bounds__(256) void wmma_gemm(
        const bf16_t* __restrict__ A, const bf16_t* __restrict__ W,
        void* __restrict__ Cout) {
    const int tid  = threadIdx.x;
    const int wave = tid >> 5;
    const int lane = tid & 31;
    const int half = lane >> 4;
    const int l16  = lane & 15;

    const int m0 = blockIdx.x * 128 + wave * 16;
    const int n0 = blockIdx.y * 64;

    v8f acc[4] = {};

    const bf16_t* arow = A + (size_t)(m0 + l16) * Dc + half * 16;
    const bf16_t* wrow = W + (size_t)(n0 + l16) * Dc + half * 16;

    for (int kk = 0; kk < Dc; kk += 32) {
        __builtin_prefetch(arow + kk + 128, 0, 1);
        v16bf a = *(const v16bf*)(arow + kk);
#pragma unroll
        for (int j = 0; j < 4; ++j) {
            v16bf bw = *(const v16bf*)(wrow + (size_t)j * 16 * Dc + kk);
            acc[j] = __builtin_amdgcn_wmma_f32_16x16x32_bf16(
                false, a, false, bw, (short)0, acc[j], false, false);
        }
    }

#pragma unroll
    for (int j = 0; j < 4; ++j) {
#pragma unroll
        for (int i = 0; i < 8; ++i) {
            int m = m0 + i + half * 8;
            int c = n0 + j * 16 + l16;
            float v = acc[j][i];
            if (MODE == 1) {
                ((float*)Cout)[(size_t)m * Dc + c] = v;
            } else {
                int b = m >> 11, s = m & (Sc - 1);
                int h = c >> 6,  d = c & 63;
                ((bf16_t*)Cout)[(((size_t)(b * Hc + h) * Sc) + s) * DHc + d] =
                    (bf16_t)v;
            }
        }
    }
}

// ---------------------------------------------------------------------------
// Flash attention (causal, online softmax)
// grid = (S/64, B*H); block = 128 threads = 4 waves
// Row sums via an extra "ones-column" WMMA; row max via DPP butterflies.
// K tile staged with GLOBAL_LOAD_ASYNC_TO_LDS_B128 (ASYNCcnt) when available.
// ---------------------------------------------------------------------------
__global__ __launch_bounds__(128) void flash_attn(
        const bf16_t* __restrict__ Q, const bf16_t* __restrict__ K,
        const bf16_t* __restrict__ V, bf16_t* __restrict__ O) {
    __shared__ bf16_t sK[32][64];        // 32 keys x dh
    __shared__ bf16_t sVt[64][32];       // dh x 32 keys (transposed V)
    __shared__ bf16_t sP[4][16][32];     // per-wave P tile

    const int bh   = blockIdx.y;
    const int b    = bh >> 4;
    const int h    = bh & 15;
    const int q0   = blockIdx.x * 64;
    const int tid  = threadIdx.x;
    const int wave = tid >> 5;
    const int lane = tid & 31;
    const int half = lane >> 4;
    const int l16  = lane & 15;

    const bf16_t* Qh = Q + (size_t)bh * Sc * DHc;
    const bf16_t* Kh = K + (size_t)bh * Sc * DHc;
    const bf16_t* Vh = V + (size_t)bh * Sc * DHc;

    const int qrow = q0 + wave * 16;
    v16bf qa0 = *(const v16bf*)(Qh + (size_t)(qrow + l16) * DHc + half * 16);
    v16bf qa1 = *(const v16bf*)(Qh + (size_t)(qrow + l16) * DHc + 32 + half * 16);

    // B-fragment with ones in column 0 only: P x ones -> row sums in col 0
    v16bf onesfrag;
    {
        bf16_t ov = (bf16_t)((l16 == 0) ? 1.0f : 0.0f);
#pragma unroll
        for (int j = 0; j < 16; ++j) onesfrag[j] = ov;
    }

    v8f acc[4] = {};
    v8f accS   = {};
    float mrow[8];
#pragma unroll
    for (int i = 0; i < 8; ++i) mrow[i] = -1e30f;

    // 1/sqrt(64) * log2(e): scores move to log2 domain, exp2 is native
    const float SLOG2E = 0.18033688011112042f;
    const int nkt = blockIdx.x * 2 + 2;

    for (int kt = 0; kt < nkt; ++kt) {
        const int kbase = kt * 32;
        // ---- cooperative stage: K tile (async if available) + V^T tile ----
        {
            int r = tid >> 2, c = (tid & 3) << 4;
            const bf16_t* kp = Kh + (size_t)(kbase + r) * DHc + c;
#if USE_ASYNC_LDS
            __builtin_amdgcn_global_load_async_to_lds_b128(
                (g_v4i_ptr)kp, (l_v4i_ptr)&sK[r][c], 0, 0);
            __builtin_amdgcn_global_load_async_to_lds_b128(
                (g_v4i_ptr)(kp + 8), (l_v4i_ptr)&sK[r][c + 8], 0, 0);
#else
            *(v16bf*)(&sK[r][c]) = *(const v16bf*)kp;
#endif
            v16bf vv = *(const v16bf*)(Vh + (size_t)(kbase + r) * DHc + c);
#pragma unroll
            for (int j = 0; j < 16; ++j) sVt[c + j][r] = vv[j];
        }
#if USE_ASYNC_LDS
        __builtin_amdgcn_s_wait_asynccnt(0);
#endif
        __syncthreads();

        // ---- S = Q K^T : two 16-key N tiles, K=64 in two wmma steps ----
        v8f sc[2];
#pragma unroll
        for (int nt = 0; nt < 2; ++nt) {
            v8f c = {};
            v16bf b0 = *(const v16bf*)(&sK[nt * 16 + l16][half * 16]);
            c = __builtin_amdgcn_wmma_f32_16x16x32_bf16(
                false, qa0, false, b0, (short)0, c, false, false);
            v16bf b1 = *(const v16bf*)(&sK[nt * 16 + l16][32 + half * 16]);
            c = __builtin_amdgcn_wmma_f32_16x16x32_bf16(
                false, qa1, false, b1, (short)0, c, false, false);
            sc[nt] = c;
        }

        // ---- scale (log2 domain) + causal mask + DPP row max ----
        const bool need_mask = (kbase + 31 > qrow);   // wave-uniform
        float mloc[8];
#pragma unroll
        for (int i = 0; i < 8; ++i) {
            float v0 = sc[0][i] * SLOG2E;
            float v1 = sc[1][i] * SLOG2E;
            if (need_mask) {
                int q = qrow + i + half * 8;
                if (kbase + l16 > q)      v0 = -1e30f;
                if (kbase + 16 + l16 > q) v1 = -1e30f;
            }
            sc[0][i] = v0;
            sc[1][i] = v1;
            mloc[i] = dpp_max16(fmaxf(v0, v1));
        }

        // ---- online softmax update: P = exp2(s - m), stage P to LDS ----
        float alpha[8];
#pragma unroll
        for (int i = 0; i < 8; ++i) {
            float mnew = fmaxf(mrow[i], mloc[i]);
            alpha[i] = exp2f(mrow[i] - mnew);
            float p0 = exp2f(sc[0][i] - mnew);
            float p1 = exp2f(sc[1][i] - mnew);
            mrow[i] = mnew;
            int m = i + half * 8;
            sP[wave][m][l16]      = (bf16_t)p0;
            sP[wave][m][16 + l16] = (bf16_t)p1;
        }
#pragma unroll
        for (int j = 0; j < 4; ++j)
#pragma unroll
            for (int i = 0; i < 8; ++i) acc[j][i] *= alpha[i];
#pragma unroll
        for (int i = 0; i < 8; ++i) accS[i] *= alpha[i];

        asm volatile("s_wait_dscnt 0x0" ::: "memory");

        // ---- O += P V  (4 dh tiles) ; row sums += P x ones ----
        v16bf pfrg = *(const v16bf*)(&sP[wave][l16][half * 16]);
#pragma unroll
        for (int j = 0; j < 4; ++j) {
            v16bf vfrg = *(const v16bf*)(&sVt[j * 16 + l16][half * 16]);
            acc[j] = __builtin_amdgcn_wmma_f32_16x16x32_bf16(
                false, pfrg, false, vfrg, (short)0, acc[j], false, false);
        }
        accS = __builtin_amdgcn_wmma_f32_16x16x32_bf16(
            false, pfrg, false, onesfrag, (short)0, accS, false, false);

        __syncthreads();
    }

    // ---- epilogue: broadcast row sums from column 0, normalize, store ----
    float linv[8];
#pragma unroll
    for (int i = 0; i < 8; ++i) {
        float l = __shfl(accS[i], 0, 16);   // col 0 of each 16-lane group
        linv[i] = 1.0f / l;
    }
#pragma unroll
    for (int j = 0; j < 4; ++j)
#pragma unroll
        for (int i = 0; i < 8; ++i) {
            int m = i + half * 8;
            int q = qrow + m;
            float val = acc[j][i] * linv[i];
            O[(size_t)(b * Sc + q) * Dc + h * DHc + j * 16 + l16] = (bf16_t)val;
        }
}

// ---------------------------------------------------------------------------
// host launcher
// ---------------------------------------------------------------------------
extern "C" void kernel_launch(void* const* d_in, const int* in_sizes, int n_in,
                              void* d_out, int out_size, void* d_ws,
                              size_t ws_size, hipStream_t stream) {
    const float* x   = (const float*)d_in[0];
    const float* WQ  = (const float*)d_in[1];
    const float* WK  = (const float*)d_in[2];
    const float* WV  = (const float*)d_in[3];
    const float* WO  = (const float*)d_in[4];
    const int*   pos = (const int*)d_in[5];

    const size_t BSD = (size_t)Bc * Sc * Dc;
    const size_t DD  = (size_t)Dc * Dc;

    char* ws = (char*)d_ws;
    size_t off = 0;
    auto alloc = [&](size_t bytes) -> void* {
        void* p = ws + off;
        off = (off + bytes + 255) & ~(size_t)255;
        return p;
    };
    bf16_t* xb = (bf16_t*)alloc(BSD * 2);
    bf16_t* wq = (bf16_t*)alloc(DD * 2);
    bf16_t* wk = (bf16_t*)alloc(DD * 2);
    bf16_t* wv = (bf16_t*)alloc(DD * 2);
    bf16_t* wo = (bf16_t*)alloc(DD * 2);
    bf16_t* Qh = (bf16_t*)alloc(BSD * 2);
    bf16_t* Kh = (bf16_t*)alloc(BSD * 2);
    bf16_t* Vh = (bf16_t*)alloc(BSD * 2);
    bf16_t* Ob = (bf16_t*)alloc(BSD * 2);

    cast_f32_bf16<<<(int)(BSD / 4 / 256), 256, 0, stream>>>(x, xb, (int)BSD);
    cast_f32_bf16<<<(int)(DD / 4 / 256), 256, 0, stream>>>(WQ, wq, (int)DD);
    cast_f32_bf16<<<(int)(DD / 4 / 256), 256, 0, stream>>>(WK, wk, (int)DD);
    cast_f32_bf16<<<(int)(DD / 4 / 256), 256, 0, stream>>>(WV, wv, (int)DD);
    cast_f32_bf16<<<(int)(DD / 4 / 256), 256, 0, stream>>>(WO, wo, (int)DD);

    dim3 ggrid(64, 16);
    wmma_gemm<0><<<ggrid, 256, 0, stream>>>(xb, wq, Qh);
    wmma_gemm<0><<<ggrid, 256, 0, stream>>>(xb, wk, Kh);
    wmma_gemm<0><<<ggrid, 256, 0, stream>>>(xb, wv, Vh);

    int rope_blocks = (Bc * Hc * Sc * 32) / 256;
    rope_kernel<<<rope_blocks, 256, 0, stream>>>(Qh, pos);
    rope_kernel<<<rope_blocks, 256, 0, stream>>>(Kh, pos);

    flash_attn<<<dim3(Sc / 64, Bc * Hc), 128, 0, stream>>>(Qh, Kh, Vh, Ob);

    wmma_gemm<1><<<ggrid, 256, 0, stream>>>(Ob, wo, d_out);
}
